// GDNBlock_71880572666624
// MI455X (gfx1250) — matrix operand order repaired and checked by reading, not compiled, not run on hardware
//
#include <hip/hip_runtime.h>
#include <stdint.h>

// Problem constants (match reference)
#define B_   4
#define S_   2048
#define D_   1024
#define H_   16
#define HD_  64
#define DFF_ 4096
#define M_   (B_*S_)   // 8192 rows

typedef __bf16 v16bf __attribute__((ext_vector_type(16)));
typedef float  v8f   __attribute__((ext_vector_type(8)));

__device__ __forceinline__ unsigned short f2bf(float f) {
  union { float f; uint32_t u; } x; x.f = f;
  uint32_t r = x.u + 0x7FFFu + ((x.u >> 16) & 1u);   // round-to-nearest-even
  return (unsigned short)(r >> 16);
}
__device__ __forceinline__ float bf2f(unsigned short h) {
  union { uint32_t u; float f; } x; x.u = ((uint32_t)h) << 16;
  return x.f;
}

// CDNA5 async global->LDS DMA (16B per lane), tracked by ASYNCcnt.
// vdst = per-lane LDS byte offset, vaddr = 64-bit global address, no saddr.
__device__ __forceinline__ void async_copy16(uint32_t lds_off, const void* gptr) {
  asm volatile("global_load_async_to_lds_b128 %0, %1, off"
               :: "v"(lds_off), "v"(gptr)
               : "memory");
}
__device__ __forceinline__ void wait_async0() {
  asm volatile("s_wait_asynccnt 0x0" ::: "memory");
}
// Flat address of a __shared__ object: low 32 bits are the LDS byte offset.
__device__ __forceinline__ uint32_t lds_off(const void* p) {
  return (uint32_t)(uintptr_t)p;
}

// ---------------------------------------------------------------------------
// Tiled transpose + fp32->bf16 cast: WT[n][k] = bf16(W[k][n]).
// Grid (N/32, K/32), 256 threads (32x8). Pre-transposing the weights lets the
// GEMM stage both A and B tiles as pure 16B DMA copies (no scalar scatter).
// ---------------------------------------------------------------------------
__global__ __launch_bounds__(256)
void transpose_cast_kernel(const float* __restrict__ W, unsigned short* __restrict__ WT,
                           int K, int N) {
  __shared__ float tile[32][33];
  const int k0 = blockIdx.y * 32, n0 = blockIdx.x * 32;
  const int tx = threadIdx.x & 31, ty = threadIdx.x >> 5;   // 32 x 8
#pragma unroll
  for (int i = 0; i < 32; i += 8)
    tile[ty + i][tx] = W[(size_t)(k0 + ty + i) * N + n0 + tx];
  __syncthreads();
#pragma unroll
  for (int i = 0; i < 32; i += 8)
    WT[(size_t)(n0 + ty + i) * K + k0 + tx] = f2bf(tile[tx][ty + i]);
}

// ---------------------------------------------------------------------------
// RMSNorm: one block per row of D=1024; output bf16
// ---------------------------------------------------------------------------
__global__ __launch_bounds__(256)
void rmsnorm_kernel(const float* __restrict__ x, const float* __restrict__ scale,
                    unsigned short* __restrict__ out) {
  __shared__ float red[8];
  const int row = blockIdx.x;
  const int tid = threadIdx.x;
  const float* xr = x + (size_t)row * D_;
  float4 v = ((const float4*)xr)[tid];                 // 256*4 = 1024
  float ss = v.x*v.x + v.y*v.y + v.z*v.z + v.w*v.w;
  for (int m = 1; m < 32; m <<= 1) ss += __shfl_xor(ss, m, 32);
  if ((tid & 31) == 0) red[tid >> 5] = ss;
  __syncthreads();
  if (tid == 0) {
    float t = 0.f;
    for (int i = 0; i < 8; i++) t += red[i];
    red[0] = rsqrtf(t * (1.0f / (float)D_) + 1e-6f);
  }
  __syncthreads();
  const float inv = red[0];
  const float4 sc = ((const float4*)scale)[tid];
  unsigned short* op = out + (size_t)row * D_ + tid * 4;
  op[0] = f2bf(v.x * inv * sc.x);
  op[1] = f2bf(v.y * inv * sc.y);
  op[2] = f2bf(v.z * inv * sc.z);
  op[3] = f2bf(v.w * inv * sc.w);
}

// ---------------------------------------------------------------------------
// WMMA GEMM: C[M,N] = A[M,K](bf16) @ WT[N,K]^T(bf16) + bias, fused epilogue.
//   FLAVOR 0: store fp32
//   FLAVOR 1: store fp32 + residual add
//   FLAVOR 2: tanh-GELU, store bf16
// Block tile 128x128, K-step 32, 256 threads = 8 waves (4 along M x 2 along N),
// each wave computes 32x64 via 2x4 v_wmma_f32_16x16x32_bf16.
// Double-buffered LDS filled by async global->LDS DMA: WMMA on tile k runs
// while ASYNCcnt-tracked copies fill tile k+1; global_prefetch warms k+2.
// M,N,K all multiples of tile sizes for this problem -> no bounds checks.
// ---------------------------------------------------------------------------
#define LDT 40   // padded LDS row stride (bf16 elems): 80B, keeps 16B alignment
template <int FLAVOR>
__global__ __launch_bounds__(256)
void gemm_bf16_wmma(const unsigned short* __restrict__ A,
                    const unsigned short* __restrict__ WT,   // [N][K]
                    const float* __restrict__ bias,
                    const float* __restrict__ Res,
                    float* __restrict__ Cf,
                    unsigned short* __restrict__ Cb,
                    int K, int N) {
  __shared__ unsigned short As[2][128 * LDT];   // [m][k]
  __shared__ unsigned short Bs[2][128 * LDT];   // [n][k]

  const int tid  = threadIdx.x;
  const int lane = tid & 31;
  const int wave = tid >> 5;
  const int wm   = wave & 3;     // wave row (0..3)
  const int wn   = wave >> 2;    // wave col (0..1)
  const int m0   = blockIdx.y * 128;
  const int n0   = blockIdx.x * 128;
  const int l15  = lane & 15;
  const int hi16 = (lane & 16) ? 1 : 0;

  // staging geometry: thread DMAs rows lr and lr+64, 8 bf16 (16B) at col lc
  const int lr = tid >> 2;            // 0..63
  const int lc = (tid & 3) << 3;      // 0,8,16,24
  const unsigned short* Aptr = A  + (size_t)(m0 + lr) * K + lc;
  const unsigned short* Bptr = WT + (size_t)(n0 + lr) * K + lc;
  const size_t rstep = (size_t)64 * K;
  const uint32_t aOff0 = lds_off(&As[0][lr * LDT + lc]);
  const uint32_t aOff1 = lds_off(&As[0][(lr + 64) * LDT + lc]);
  const uint32_t bOff0 = lds_off(&Bs[0][lr * LDT + lc]);
  const uint32_t bOff1 = lds_off(&Bs[0][(lr + 64) * LDT + lc]);
  const uint32_t bufStride = 128 * LDT * sizeof(unsigned short);   // 10240 B

  v8f acc[2][4];
#pragma unroll
  for (int i = 0; i < 2; i++)
#pragma unroll
    for (int j = 0; j < 4; j++)
      acc[i][j] = (v8f){0.f,0.f,0.f,0.f,0.f,0.f,0.f,0.f};

  // prologue: DMA tile 0 into buffer 0
  async_copy16(aOff0, Aptr);
  async_copy16(aOff1, Aptr + rstep);
  async_copy16(bOff0, Bptr);
  async_copy16(bOff1, Bptr + rstep);
  wait_async0();
  __syncthreads();

  int cur = 0;
  for (int k0 = 0; k0 < K; k0 += 32) {
    const bool notlast = (k0 + 32 < K);
    if (notlast) {
      const uint32_t bs = (uint32_t)(cur ^ 1) * bufStride;
      async_copy16(aOff0 + bs, Aptr + k0 + 32);
      async_copy16(aOff1 + bs, Aptr + rstep + k0 + 32);
      async_copy16(bOff0 + bs, Bptr + k0 + 32);
      async_copy16(bOff1 + bs, Bptr + rstep + k0 + 32);
      if (k0 + 64 < K) {   // warm GL2/WGP$ for the tile after next
        __builtin_prefetch(Aptr + k0 + 64, 0, 1);
        __builtin_prefetch(Bptr + k0 + 64, 0, 1);
      }
    }

    // A fragment (16x32 bf16): lanes 0-15 row M=l15 hold K {0..7,16..23},
    // lanes 16-31 hold K {8..15,24..31}.
    v16bf afrag[2], bfrag[4];
#pragma unroll
    for (int tm = 0; tm < 2; tm++) {
      int r  = wm * 32 + tm * 16 + l15;
      int lo = hi16 ? 8 : 0;
      union { uint4 u[2]; v16bf v; } t;
      t.u[0] = *(const uint4*)(&As[cur][r * LDT + lo]);
      t.u[1] = *(const uint4*)(&As[cur][r * LDT + lo + 16]);
      afrag[tm] = t.v;
    }
    // B fragment (32x16 bf16): lane col N=l15; lanes 0-15 K {0..7,16..23}, 16-31 K {8..15,24..31}.
#pragma unroll
    for (int tn = 0; tn < 4; tn++) {
      int n   = wn * 64 + tn * 16 + l15;
      int klo = hi16 ? 8 : 0;
      union { uint4 u[2]; v16bf v; } t;
      t.u[0] = *(const uint4*)(&Bs[cur][n * LDT + klo]);
      t.u[1] = *(const uint4*)(&Bs[cur][n * LDT + klo + 16]);
      bfrag[tn] = t.v;
    }

#pragma unroll
    for (int tm = 0; tm < 2; tm++)
#pragma unroll
      for (int tn = 0; tn < 4; tn++)
        acc[tm][tn] = __builtin_amdgcn_wmma_f32_16x16x32_bf16(
            false, afrag[tm], false, bfrag[tn], (short)0, acc[tm][tn], false, false);

    if (notlast) wait_async0();
    __syncthreads();
    cur ^= 1;
  }

  // epilogue: acc VGPR r holds row (base+r) lanes 0-15 / (base+8+r) lanes 16-31
  const int rofs = hi16 ? 8 : 0;
  const int colBase = n0 + wn * 64 + l15;
  const int rowBase = m0 + wm * 32 + rofs;
#pragma unroll
  for (int tm = 0; tm < 2; tm++) {
#pragma unroll
    for (int tn = 0; tn < 4; tn++) {
      const int col = colBase + tn * 16;
      const float bia = bias[col];
      size_t idx = (size_t)(rowBase + tm * 16) * N + col;
#pragma unroll
      for (int rr = 0; rr < 8; rr++) {
        float v = acc[tm][tn][rr] + bia;
        if (FLAVOR == 0) {
          Cf[idx] = v;
        } else if (FLAVOR == 1) {
          Cf[idx] = v + Res[idx];
        } else {
          float g = 0.5f * v * (1.0f + tanhf(0.7978845608028654f * (v + 0.044715f * v * v * v)));
          Cb[idx] = f2bf(g);
        }
        idx += (size_t)N;
      }
    }
  }
}

// ---------------------------------------------------------------------------
// Per-row: beta = sigmoid(h @ wbeta + bbeta) (GEMV, H=16) and normalize k per head.
// One block per row (B*S rows), 256 threads: 16 threads per head.
// ---------------------------------------------------------------------------
__global__ __launch_bounds__(256)
void knorm_beta_kernel(const unsigned short* __restrict__ H1,
                       const float* __restrict__ WB, const float* __restrict__ BB,
                       float* __restrict__ Kbuf, float* __restrict__ Beta) {
  __shared__ float hrow[D_];
  const int row = blockIdx.x;
  const int tid = threadIdx.x;
  for (int i = tid; i < D_; i += 256) hrow[i] = bf2f(H1[(size_t)row * D_ + i]);
  __syncthreads();
  const int hh = tid >> 4, j = tid & 15;
  // beta GEMV
  float p = 0.f;
#pragma unroll 4
  for (int i = 0; i < 64; i++) {
    int d = j + (i << 4);
    p += hrow[d] * WB[d * H_ + hh];
  }
  p += __shfl_xor(p, 1, 32); p += __shfl_xor(p, 2, 32);
  p += __shfl_xor(p, 4, 32); p += __shfl_xor(p, 8, 32);
  if (j == 0) Beta[(size_t)row * H_ + hh] = 1.f / (1.f + expf(-(p + BB[hh])));
  // k normalization over head_dim=64 (4 elems/thread, 16 threads/head)
  float kv[4]; float s = 0.f;
  const size_t base = (size_t)row * D_ + hh * HD_ + j * 4;
#pragma unroll
  for (int l = 0; l < 4; l++) { kv[l] = Kbuf[base + l]; s += kv[l] * kv[l]; }
  s += __shfl_xor(s, 1, 32); s += __shfl_xor(s, 2, 32);
  s += __shfl_xor(s, 4, 32); s += __shfl_xor(s, 8, 32);
  const float inv = 1.0f / (sqrtf(s) + 1e-6f);
#pragma unroll
  for (int l = 0; l < 4; l++) Kbuf[base + l] = kv[l] * inv;
}

// ---------------------------------------------------------------------------
// GDN delta-rule scan. One block per (b,h); 64x64 fp32 state in registers:
// thread = (row d = tid/4, K-chunk c = tid%4 covering 16 cols). Sequential S loop.
// Output written as bf16 (feeds wo WMMA GEMM).
// ---------------------------------------------------------------------------
__global__ __launch_bounds__(256)
void gdn_scan_kernel(const float* __restrict__ Q, const float* __restrict__ Kb,
                     const float* __restrict__ V, const float* __restrict__ Beta,
                     unsigned short* __restrict__ O) {
  __shared__ float ks[HD_], qs[HD_], vs[HD_], red[HD_];
  __shared__ float betaS;
  const int b = blockIdx.x / H_;
  const int h = blockIdx.x % H_;
  const int tid = threadIdx.x;
  const int d = tid >> 2, c = tid & 3, e0 = c << 4;

  float Sreg[16];
#pragma unroll
  for (int i = 0; i < 16; i++) Sreg[i] = 0.f;

  for (int t = 0; t < S_; t++) {
    const size_t base = ((size_t)(b * S_ + t)) * D_ + h * HD_;
    if (tid < HD_) { ks[tid] = Kb[base + tid]; qs[tid] = Q[base + tid]; vs[tid] = V[base + tid]; }
    if (tid == 0) betaS = Beta[(size_t)(b * S_ + t) * H_ + h];
    __syncthreads();
    // Sk = S @ k
    float p = 0.f;
#pragma unroll
    for (int i = 0; i < 16; i++) p += Sreg[i] * ks[e0 + i];
    p += __shfl_xor(p, 1, 32); p += __shfl_xor(p, 2, 32);
    if (c == 0) red[d] = p;
    __syncthreads();
    // delta-rule update + o = S @ q
    const float bd = betaS * (vs[d] - red[d]);
    float p2 = 0.f;
#pragma unroll
    for (int i = 0; i < 16; i++) {
      Sreg[i] += bd * ks[e0 + i];
      p2 += Sreg[i] * qs[e0 + i];
    }
    p2 += __shfl_xor(p2, 1, 32); p2 += __shfl_xor(p2, 2, 32);
    if (c == 0) O[base + d] = f2bf(p2);
    __syncthreads();
  }
}

// ---------------------------------------------------------------------------
// Orchestration
// ---------------------------------------------------------------------------
extern "C" void kernel_launch(void* const* d_in, const int* in_sizes, int n_in,
                              void* d_out, int out_size, void* d_ws, size_t ws_size,
                              hipStream_t stream) {
  const float* x     = (const float*)d_in[0];
  const float* wq    = (const float*)d_in[1];
  const float* bq    = (const float*)d_in[2];
  const float* wk    = (const float*)d_in[3];
  const float* bk    = (const float*)d_in[4];
  const float* wv    = (const float*)d_in[5];
  const float* bv    = (const float*)d_in[6];
  const float* wbeta = (const float*)d_in[7];
  const float* bbeta = (const float*)d_in[8];
  const float* wo    = (const float*)d_in[9];
  const float* bo    = (const float*)d_in[10];
  const float* w1    = (const float*)d_in[11];
  const float* b1    = (const float*)d_in[12];
  const float* w2    = (const float*)d_in[13];
  const float* b2    = (const float*)d_in[14];
  const float* ln1   = (const float*)d_in[15];
  const float* ln2   = (const float*)d_in[16];
  float* out = (float*)d_out;

  // --- workspace carve (with buffer reuse) ---
  char* ws = (char*)d_ws;
  auto carve = [&](size_t bytes) -> char* {
    char* p = ws;
    ws += (bytes + 255) & ~(size_t)255;
    return p;
  };
  const size_t MD = (size_t)M_ * D_;
  char* qkv_blob = carve(3 * MD * sizeof(float));          // q,k,v fp32; later a1 (bf16, 64MB <= 96MB)
  float* qbuf = (float*)qkv_blob;
  float* kbuf = qbuf + MD;
  float* vbuf = kbuf + MD;
  unsigned short* a1 = (unsigned short*)qkv_blob;          // reuse after scan
  unsigned short* h1 = (unsigned short*)carve(MD * 2);     // rmsnorm1 out; reused as h2
  unsigned short* h2 = h1;
  unsigned short* attn = (unsigned short*)carve(MD * 2);   // scan output bf16
  float* x2   = (float*)carve(MD * sizeof(float));         // post-attention residual stream
  float* beta = (float*)carve((size_t)M_ * H_ * sizeof(float));
  unsigned short* wq_bf = (unsigned short*)carve((size_t)D_ * D_ * 2);   // all transposed [N][K]
  unsigned short* wk_bf = (unsigned short*)carve((size_t)D_ * D_ * 2);
  unsigned short* wv_bf = (unsigned short*)carve((size_t)D_ * D_ * 2);
  unsigned short* wo_bf = (unsigned short*)carve((size_t)D_ * D_ * 2);
  unsigned short* w1_bf = (unsigned short*)carve((size_t)D_ * DFF_ * 2);
  unsigned short* w2_bf = (unsigned short*)carve((size_t)DFF_ * D_ * 2);
  (void)ws_size; (void)n_in; (void)in_sizes; (void)out_size;

  // 1) weight transpose-casts fp32 [K][N] -> bf16 [N][K]
  dim3 tDD(D_ / 32, D_ / 32);
  transpose_cast_kernel<<<tDD, 256, 0, stream>>>(wq, wq_bf, D_, D_);
  transpose_cast_kernel<<<tDD, 256, 0, stream>>>(wk, wk_bf, D_, D_);
  transpose_cast_kernel<<<tDD, 256, 0, stream>>>(wv, wv_bf, D_, D_);
  transpose_cast_kernel<<<tDD, 256, 0, stream>>>(wo, wo_bf, D_, D_);
  dim3 t1(DFF_ / 32, D_ / 32);
  transpose_cast_kernel<<<t1, 256, 0, stream>>>(w1, w1_bf, D_, DFF_);
  dim3 t2(D_ / 32, DFF_ / 32);
  transpose_cast_kernel<<<t2, 256, 0, stream>>>(w2, w2_bf, DFF_, D_);

  // 2) h1 = rmsnorm(x) * ln1  (bf16)
  rmsnorm_kernel<<<M_, 256, 0, stream>>>(x, ln1, h1);

  // 3-5) Q, K, V projections (WMMA, fp32 out)
  dim3 gD(D_ / 128, M_ / 128);
  gemm_bf16_wmma<0><<<gD, 256, 0, stream>>>(h1, wq_bf, bq, nullptr, qbuf, nullptr, D_, D_);
  gemm_bf16_wmma<0><<<gD, 256, 0, stream>>>(h1, wk_bf, bk, nullptr, kbuf, nullptr, D_, D_);
  gemm_bf16_wmma<0><<<gD, 256, 0, stream>>>(h1, wv_bf, bv, nullptr, vbuf, nullptr, D_, D_);

  // 6) k-normalization per head + beta = sigmoid(h1 @ wbeta + bbeta)
  knorm_beta_kernel<<<M_, 256, 0, stream>>>(h1, wbeta, bbeta, kbuf, beta);

  // 7) delta-rule recurrent scan -> attn (bf16)
  gdn_scan_kernel<<<B_ * H_, 256, 0, stream>>>(qbuf, kbuf, vbuf, beta, attn);

  // 8) x2 = attn @ wo + bo + x  (WMMA, residual fused)
  gemm_bf16_wmma<1><<<gD, 256, 0, stream>>>(attn, wo_bf, bo, x, x2, nullptr, D_, D_);

  // 9) h2 = rmsnorm(x2) * ln2  (bf16)
  rmsnorm_kernel<<<M_, 256, 0, stream>>>(x2, ln2, h2);

  // 10) a1 = gelu(h2 @ w1 + b1)  (WMMA, bf16 out)   [a1 reuses q/k/v space]
  dim3 gF(DFF_ / 128, M_ / 128);
  gemm_bf16_wmma<2><<<gF, 256, 0, stream>>>(h2, w1_bf, b1, nullptr, nullptr, a1, D_, DFF_);

  // 11) out = a1 @ w2 + b2 + x2  (WMMA, residual fused, fp32 out)
  gemm_bf16_wmma<1><<<gD, 256, 0, stream>>>(a1, w2_bf, b2, x2, out, nullptr, DFF_, D_);
}
